// LaCTSWIGLULayer_18245021073758
// MI455X (gfx1250) — compile-verified, matrix operand order checked by reading, not run on hardware
//
#include <hip/hip_runtime.h>
#include <hip/hip_bf16.h>

// ---------------------------------------------------------------------------
// LaCT SWIGLU layer for MI455X (gfx1250): all matmuls via v_wmma_f32_16x16x32_bf16,
// GEMM tiles staged to LDS via the Tensor Data Mover when available.
// B=1, S=4096, D=1024, NH=16, HD=64, FH=4, FD=DH=256, CS=1024, NC=4
// ---------------------------------------------------------------------------

typedef unsigned short u16;
typedef __attribute__((ext_vector_type(16))) __bf16 v16bf;
typedef __attribute__((ext_vector_type(8)))  float  v8f;
typedef __attribute__((ext_vector_type(4)))  unsigned int v4u_t;
typedef __attribute__((ext_vector_type(8)))  int v8i_t;
typedef __attribute__((ext_vector_type(4)))  int v4i_t;

#define S_LEN 4096
#define D_DIM 1024
#define NH_N  16
#define HD_N  64
#define FH_N  4
#define FD_N  256
#define CS_N  1024
#define NC_N  4
#define BASE_LR_INV -6.907255f
#define NEG_BIG -3.0e38f

// --- Tensor Data Mover availability (clang-23 6-arg builtin form) ----------
#if defined(__has_builtin)
#if __has_builtin(__builtin_amdgcn_tensor_load_to_lds) && __has_builtin(__builtin_amdgcn_s_wait_tensorcnt)
#define TDM_AVAIL 1
#endif
#endif
#ifndef TDM_AVAIL
#define TDM_AVAIL 0
#endif

__device__ __forceinline__ u16 f2bf(float f) {
  unsigned u = __float_as_uint(f);
  unsigned r = u + 0x7FFFu + ((u >> 16) & 1u);
  return (u16)(r >> 16);
}
__device__ __forceinline__ float bf2f(u16 b) {
  return __uint_as_float(((unsigned)b) << 16);
}
__device__ __forceinline__ __bf16 bfbits(u16 b) {
  union { u16 s; __bf16 h; } c; c.s = b; return c.h;
}
__device__ __forceinline__ float sigmoidf_(float x) {
  return 1.0f / (1.0f + __expf(-x));
}

#if TDM_AVAIL
// Async 2D tile load (rows x cols bf16, row stride lda elems) into LDS with
// TDM padding: 4 DWORDs inserted after every 16 DWORDs (64B row -> 80B LDS row,
// i.e. our G_LDS=40 u16 padded layout). D# layout per CDNA5 ISA 8.3/8.4.
__device__ __forceinline__ void tdm_load_tile_2d(unsigned lds_addr, const u16* gptr,
                                                 int lda_elems, int rows, int cols) {
  unsigned long long ga = (unsigned long long)(const void*)gptr;
  v4u_t g0;
  g0[0] = 1u;                                           // count=1, user descriptor
  g0[1] = lds_addr;                                     // lds_addr[31:0]
  g0[2] = (unsigned)(ga & 0xFFFFFFFFull);               // global_addr[31:0]
  g0[3] = ((unsigned)(ga >> 32) & 0x01FFFFFFu) | (2u << 30);  // addr[56:32] | type=2
  v8i_t g1;
  g1[0] = (int)((1u << 16) | (1u << 20) | (3u << 22) | (3u << 25));
         // data_size=2B | pad_enable | pad_interval=16DW | pad_amount=4DW
  g1[1] = 0;                                            // tensor_dim0 = 1<<30 (low16=0)
  g1[2] = (int)0x4000u;                                 // tensor_dim0 hi | tensor_dim1 lo16
  g1[3] = (int)(0x4000u | ((unsigned)cols << 16));      // tensor_dim1 hi | tile_dim0
  g1[4] = rows;                                         // tile_dim1 | tile_dim2=0
  g1[5] = lda_elems;                                    // tensor_dim0_stride[31:0]
  g1[6] = 0;
  g1[7] = 0;
  v4i_t z4 = {0, 0, 0, 0};
  v8i_t z8 = {0, 0, 0, 0, 0, 0, 0, 0};
  __builtin_amdgcn_tensor_load_to_lds(g0, g1, z4, z4, z8, 0);
}
#endif

// ---------------------------------------------------------------------------
// Generic batched bf16 GEMM: C[m,n] = sum_k opA[m,k]*opB[n,k] (+ C if betaOne)
// TA/TB template: operand stored [M,K] (0) or [K,M] (1, transposed at stage).
// Tile 64x64xK32, 4 waves, each wave computes a 32x32 quadrant (2x2 WMMA frags).
// All dims used are multiples of 64/32, lda/ldb multiples of 8 (16B alignment).
// ---------------------------------------------------------------------------
#define G_TM 64
#define G_TN 64
#define G_TK 32
#define G_LDS 40   // padded u16 row stride (80B, matches TDM pad config)

template <int TA, int TB>
__global__ __launch_bounds__(128) void gemm_bf16_wmma(
    const u16* __restrict__ A, const u16* __restrict__ B, float* __restrict__ C,
    int M, int N, int K, int lda, int ldb, int ldc,
    long sA, long sB, long sC, int betaOne)
{
  __shared__ u16 As[G_TM * G_LDS];
  __shared__ u16 Bs[G_TN * G_LDS];
  const int bz = blockIdx.z;
  A += (long)bz * sA; B += (long)bz * sB; C += (long)bz * sC;
  const int tid  = threadIdx.x;
  const int wave = tid >> 5;
  const int lane = tid & 31;
  const int half = lane >> 4;
  const int l16  = lane & 15;
  const int tm0 = blockIdx.y * G_TM;
  const int tn0 = blockIdx.x * G_TN;
  const int m0 = tm0 + (wave >> 1) * 32;
  const int n0 = tn0 + (wave & 1) * 32;

  v8f acc[2][2];
  if (betaOne) {
    for (int i = 0; i < 2; ++i)
      for (int j = 0; j < 2; ++j) {
        const int mb = m0 + i * 16 + half * 8;
        const int nb = n0 + j * 16 + l16;
        for (int r = 0; r < 8; ++r)
          acc[i][j][r] = C[(long)(mb + r) * ldc + nb];
      }
  } else {
    for (int i = 0; i < 2; ++i)
      for (int j = 0; j < 2; ++j)
        for (int r = 0; r < 8; ++r) acc[i][j][r] = 0.0f;
  }

#if TDM_AVAIL
  const unsigned as_base = (unsigned)(unsigned long long)(const void*)&As[0];
  const unsigned bs_base = (unsigned)(unsigned long long)(const void*)&Bs[0];
#endif

  for (int k0 = 0; k0 < K; k0 += G_TK) {
#if TDM_AVAIL
    // Non-transposed tiles move via the Tensor Data Mover (wave 0 issues).
    if (wave == 0) {
      if (TA == 0) tdm_load_tile_2d(as_base, A + (long)tm0 * lda + k0, lda, G_TM, G_TK);
      if (TB == 0) tdm_load_tile_2d(bs_base, B + (long)tn0 * ldb + k0, ldb, G_TN, G_TK);
      if (TA == 0 || TB == 0) __builtin_amdgcn_s_wait_tensorcnt(0);
    }
#else
    if (TA == 0) {
      for (int i = tid; i < G_TM * 4; i += 128) {       // 4 x b128 per 32-elem row
        const int r = i >> 2, cc = (i & 3) * 8;
        const uint4 v_ = *(const uint4*)(A + (long)(tm0 + r) * lda + k0 + cc);
        *(uint4*)&As[r * G_LDS + cc] = v_;
      }
    }
    if (TB == 0) {
      for (int i = tid; i < G_TN * 4; i += 128) {
        const int r = i >> 2, cc = (i & 3) * 8;
        const uint4 v_ = *(const uint4*)(B + (long)(tn0 + r) * ldb + k0 + cc);
        *(uint4*)&Bs[r * G_LDS + cc] = v_;
      }
    }
#endif
    if (TA == 1) {  // A stored [K,M]: vector-load along m, scatter into LDS
      for (int i = tid; i < G_TK * 8; i += 128) {
        const int kk = i >> 3, cc = (i & 7) * 8;
        const uint4 v_ = *(const uint4*)(A + (long)(k0 + kk) * lda + tm0 + cc);
        const u16* pv = (const u16*)&v_;
        for (int j = 0; j < 8; ++j) As[(cc + j) * G_LDS + kk] = pv[j];
      }
    }
    if (TB == 1) {
      for (int i = tid; i < G_TK * 8; i += 128) {
        const int kk = i >> 3, cc = (i & 7) * 8;
        const uint4 v_ = *(const uint4*)(B + (long)(k0 + kk) * ldb + tn0 + cc);
        const u16* pv = (const u16*)&v_;
        for (int j = 0; j < 8; ++j) Bs[(cc + j) * G_LDS + kk] = pv[j];
      }
    }
    if (k0 + G_TK < K) {
      __builtin_prefetch(A + (long)(TA ? (k0 + G_TK) : tm0) * lda, 0, 1);
      __builtin_prefetch(B + (long)(TB ? (k0 + G_TK) : tn0) * ldb, 0, 1);
    }
    __syncthreads();

    v16bf afr[2], bfr[2];
    const int kb = half * 8;
    for (int i = 0; i < 2; ++i) {
      const u16* ar = &As[((wave >> 1) * 32 + i * 16 + l16) * G_LDS];
      for (int j = 0; j < 8; ++j) {
        afr[i][j]     = bfbits(ar[kb + j]);
        afr[i][8 + j] = bfbits(ar[16 + kb + j]);
      }
    }
    for (int i = 0; i < 2; ++i) {
      const u16* br = &Bs[((wave & 1) * 32 + i * 16 + l16) * G_LDS];
      for (int j = 0; j < 8; ++j) {
        bfr[i][j]     = bfbits(br[kb + j]);
        bfr[i][8 + j] = bfbits(br[16 + kb + j]);
      }
    }
    for (int i = 0; i < 2; ++i)
      for (int j = 0; j < 2; ++j)
        acc[i][j] = __builtin_amdgcn_wmma_f32_16x16x32_bf16(
            false, afr[i], false, bfr[j], (short)0, acc[i][j], false, false);
    __syncthreads();
  }

  for (int i = 0; i < 2; ++i)
    for (int j = 0; j < 2; ++j) {
      const int mb = m0 + i * 16 + half * 8;
      const int nb = n0 + j * 16 + l16;
      for (int r = 0; r < 8; ++r)
        C[(long)(mb + r) * ldc + nb] = acc[i][j][r];
    }
}

// ---------------------------------------------------------------------------
// fp32 -> bf16 convert
// ---------------------------------------------------------------------------
__global__ void cvt_f32_bf16_kernel(const float* __restrict__ in, u16* __restrict__ out, long n) {
  long i = (long)blockIdx.x * blockDim.x + threadIdx.x;
  if (i < n) out[i] = f2bf(in[i]);
}

// ---------------------------------------------------------------------------
// RMSNorm in place over `cols` of each row (row stride ld)
// ---------------------------------------------------------------------------
__global__ __launch_bounds__(256) void rmsnorm_kernel(
    float* __restrict__ base, const float* __restrict__ w, int ld, int cols)
{
  __shared__ float red[8];
  float* row = base + (long)blockIdx.x * ld;
  float ss = 0.f;
  for (int c = threadIdx.x; c < cols; c += 256) { float v_ = row[c]; ss += v_ * v_; }
  for (int o = 16; o > 0; o >>= 1) ss += __shfl_xor(ss, o, 32);
  if ((threadIdx.x & 31) == 0) red[threadIdx.x >> 5] = ss;
  __syncthreads();
  if (threadIdx.x == 0) {
    float t = 0.f; for (int i = 0; i < 8; ++i) t += red[i];
    red[0] = rsqrtf(t / cols + 1e-5f);
  }
  __syncthreads();
  const float rs = red[0];
  for (int c = threadIdx.x; c < cols; c += 256) row[c] = row[c] * rs * w[c];
}

// ---------------------------------------------------------------------------
// Fast-weight path prep: silu(+affine)(+l2norm) -> bf16 [FH, S, FD]
// ---------------------------------------------------------------------------
__global__ __launch_bounds__(256) void fw_kernel(
    const float* __restrict__ in, int ldin,
    const float* __restrict__ scale, const float* __restrict__ offset,
    int do_l2, u16* __restrict__ out)
{
  __shared__ float red[8];
  const int s = blockIdx.x, fh = blockIdx.y, d = threadIdx.x;
  const int col = fh * FD_N + d;
  float x = in[(long)s * ldin + col];
  float f = x * sigmoidf_(x);
  if (scale) f = f * scale[col * 2] + offset[col * 2];
  float nrm = 1.f;
  if (do_l2) {
    float ss = f * f;
    for (int o = 16; o > 0; o >>= 1) ss += __shfl_xor(ss, o, 32);
    if ((d & 31) == 0) red[d >> 5] = ss;
    __syncthreads();
    if (d == 0) { float t = 0.f; for (int i = 0; i < 8; ++i) t += red[i]; red[0] = rsqrtf(t + 1e-6f); }
    __syncthreads();
    nrm = red[0];
  }
  out[((long)fh * S_LEN + s) * FD_N + d] = f2bf(f * nrm);
}

// ---------------------------------------------------------------------------
// Per-token learning rates: lr[s, j] = softplus(x[s].W_lr[j] + b[j] + base)
// ---------------------------------------------------------------------------
__global__ __launch_bounds__(256) void lr_kernel(
    const float* __restrict__ x, const float* __restrict__ Wlr,
    const float* __restrict__ blr, float* __restrict__ lr)
{
  __shared__ float red[8];
  const int s = blockIdx.x;
  const float* xr = x + (long)s * D_DIM;
  for (int j = 0; j < 12; ++j) {
    const float* wr = Wlr + (long)j * D_DIM;
    float ss = 0.f;
    for (int c = threadIdx.x; c < D_DIM; c += 256) ss += xr[c] * wr[c];
    for (int o = 16; o > 0; o >>= 1) ss += __shfl_xor(ss, o, 32);
    if ((threadIdx.x & 31) == 0) red[threadIdx.x >> 5] = ss;
    __syncthreads();
    if (threadIdx.x == 0) {
      float t = 0.f; for (int i = 0; i < 8; ++i) t += red[i];
      t += blr[j] + BASE_LR_INV;
      lr[(long)s * 12 + j] = (t > 20.f) ? t : log1pf(__expf(t));
    }
    __syncthreads();
  }
}

// ---------------------------------------------------------------------------
// RoPE (optional) + head-major transpose: fp32 [S, NH*HD]@ld -> bf16 [NH, S, HD]
// ---------------------------------------------------------------------------
__global__ void rope_transpose_kernel(
    const float* __restrict__ in, u16* __restrict__ out, int do_rope, int ldin)
{
  long idx = (long)blockIdx.x * blockDim.x + threadIdx.x; // over S*NH*(HD/2)
  const long total = (long)S_LEN * NH_N * (HD_N / 2);
  if (idx >= total) return;
  const int p = (int)(idx % (HD_N / 2));
  const int h = (int)((idx / (HD_N / 2)) % NH_N);
  const int s = (int)(idx / ((long)(HD_N / 2) * NH_N));
  float x1 = in[(long)s * ldin + h * HD_N + p];
  float x2 = in[(long)s * ldin + h * HD_N + p + HD_N / 2];
  float o1 = x1, o2 = x2;
  if (do_rope) {
    const float inv = __powf(500000.0f, -(2.0f * p) / (float)HD_N);
    const float ang = (float)s * inv;
    float sn, cv;
    __sincosf(ang, &sn, &cv);
    o1 = x1 * cv - x2 * sn;
    o2 = x2 * cv + x1 * sn;
  }
  u16* o = out + ((long)h * S_LEN + s) * HD_N;
  o[p] = f2bf(o1);
  o[p + HD_N / 2] = f2bf(o2);
}

// ---------------------------------------------------------------------------
// Sliding-window flash attention. Q/K/V bf16 [NH, S, HD]. Block = 64 query rows
// of one (head, chunk); each of 4 waves owns 16 rows. Online softmax; both
// QK^T and P.V via WMMA. Output fp32 [S, D] interleaved (col = h*HD + d).
// ---------------------------------------------------------------------------
__global__ __launch_bounds__(128) void attn_kernel(
    const u16* __restrict__ Q, const u16* __restrict__ Kg,
    const u16* __restrict__ V, float* __restrict__ ao)
{
  __shared__ u16 Kt[64 * 72];      // [key][hd]
  __shared__ u16 Vt[64 * 72];      // [hd][key]  (transposed stage)
  __shared__ u16 Pt[4][16 * 72];   // per-wave P tile [row][key]
  const int h = blockIdx.z;
  const int c = blockIdx.y;
  const int mt = blockIdx.x;
  const int tid = threadIdx.x, wave = tid >> 5, lane = tid & 31;
  const int half = lane >> 4, l16 = lane & 15;
  const int rowbase = c * CS_N + mt * 64 + wave * 16;
  const u16* Qh = Q  + (long)h * S_LEN * HD_N;
  const u16* Kh = Kg + (long)h * S_LEN * HD_N;
  const u16* Vh = V  + (long)h * S_LEN * HD_N;

  // Q fragments for this wave's 16 rows (HD=64 -> two 32-wide k-steps),
  // vectorized: 4x b128 per lane.
  v16bf qa[2];
  {
    const u16* qr = Qh + (long)(rowbase + l16) * HD_N;
    const int kb = half * 8;
    for (int f = 0; f < 2; ++f) {
      const uint4 lo = *(const uint4*)(qr + f * 32 + kb);
      const uint4 hi = *(const uint4*)(qr + f * 32 + 16 + kb);
      const u16* plo = (const u16*)&lo;
      const u16* phi = (const u16*)&hi;
      for (int j = 0; j < 8; ++j) {
        qa[f][j]     = bfbits(plo[j]);
        qa[f][8 + j] = bfbits(phi[j]);
      }
    }
  }

  float mrow[8], lrow[8];
  v8f oacc[4];
  for (int r = 0; r < 8; ++r) { mrow[r] = NEG_BIG; lrow[r] = 0.f; }
  for (int nb = 0; nb < 4; ++nb)
    for (int r = 0; r < 8; ++r) oacc[nb][r] = 0.f;

  int kstart = c * CS_N - CS_N; if (kstart < 0) kstart = 0;
  const int kend = c * CS_N + mt * 64 + 64;

  for (int j0 = kstart; j0 < kend; j0 += 64) {
    // stage K ([key][hd], vector) and V transposed ([hd][key], scatter)
    for (int i = tid; i < 64 * 8; i += 128) {
      const int kk = i >> 3, cc = (i & 7) * 8;
      const uint4 kv = *(const uint4*)(Kh + (long)(j0 + kk) * HD_N + cc);
      *(uint4*)&Kt[kk * 72 + cc] = kv;
      const uint4 vv = *(const uint4*)(Vh + (long)(j0 + kk) * HD_N + cc);
      const u16* pv = (const u16*)&vv;
      for (int j = 0; j < 8; ++j) Vt[(cc + j) * 72 + kk] = pv[j];
    }
    __syncthreads();

    // S tile (16 rows x 64 keys)
    v8f sfr[4];
    const int kb = half * 8;
    for (int nb = 0; nb < 4; ++nb) {
      v16bf kb0, kb1;
      const u16* kr = &Kt[(nb * 16 + l16) * 72];
      for (int j = 0; j < 8; ++j) {
        kb0[j] = bfbits(kr[kb + j]);      kb0[8 + j] = bfbits(kr[16 + kb + j]);
        kb1[j] = bfbits(kr[32 + kb + j]); kb1[8 + j] = bfbits(kr[48 + kb + j]);
      }
      v8f z; for (int r = 0; r < 8; ++r) z[r] = 0.f;
      z = __builtin_amdgcn_wmma_f32_16x16x32_bf16(false, qa[0], false, kb0, (short)0, z, false, false);
      z = __builtin_amdgcn_wmma_f32_16x16x32_bf16(false, qa[1], false, kb1, (short)0, z, false, false);
      sfr[nb] = z;
    }

    // mask + scale + block row-max
    const float sc = 0.125f; // 64^-0.5
    float bmax[8];
    for (int r = 0; r < 8; ++r) bmax[r] = NEG_BIG;
    for (int nb = 0; nb < 4; ++nb) {
      const int kpos = j0 + nb * 16 + l16;
      for (int r = 0; r < 8; ++r) {
        const int g = rowbase + half * 8 + r;
        float v_ = sfr[nb][r] * sc;
        const bool ok = (kpos <= g) && (kpos > g - CS_N);
        v_ = ok ? v_ : NEG_BIG;
        sfr[nb][r] = v_;
        bmax[r] = fmaxf(bmax[r], v_);
      }
    }
    for (int r = 0; r < 8; ++r)
      for (int o = 1; o < 16; o <<= 1)
        bmax[r] = fmaxf(bmax[r], __shfl_xor(bmax[r], o, 32));

    float alpha[8];
    for (int r = 0; r < 8; ++r) {
      const float mn = fmaxf(mrow[r], bmax[r]);
      alpha[r] = __expf(mrow[r] - mn);
      mrow[r] = mn;
      lrow[r] *= alpha[r];
    }
    float rsum[8];
    for (int r = 0; r < 8; ++r) rsum[r] = 0.f;
    for (int nb = 0; nb < 4; ++nb)
      for (int r = 0; r < 8; ++r) {
        float p = __expf(sfr[nb][r] - mrow[r]);
        p = (sfr[nb][r] <= -1.0e37f) ? 0.f : p;
        sfr[nb][r] = p;
        rsum[r] += p;
      }
    for (int r = 0; r < 8; ++r)
      for (int o = 1; o < 16; o <<= 1)
        rsum[r] += __shfl_xor(rsum[r], o, 32);
    for (int r = 0; r < 8; ++r) lrow[r] += rsum[r];
    for (int nb = 0; nb < 4; ++nb)
      for (int r = 0; r < 8; ++r) oacc[nb][r] *= alpha[r];

    // C-layout P -> row-major LDS -> A-fragments
    u16* pw = Pt[wave];
    for (int nb = 0; nb < 4; ++nb)
      for (int r = 0; r < 8; ++r)
        pw[(half * 8 + r) * 72 + nb * 16 + l16] = f2bf(sfr[nb][r]);

    v16bf pa[2];
    {
      const u16* pr = pw + l16 * 72;
      for (int f = 0; f < 2; ++f)
        for (int j = 0; j < 8; ++j) {
          pa[f][j]     = bfbits(pr[f * 32 + kb + j]);
          pa[f][8 + j] = bfbits(pr[f * 32 + 16 + kb + j]);
        }
    }
    for (int nb = 0; nb < 4; ++nb) {
      v16bf vb0, vb1;
      const u16* vr = &Vt[(nb * 16 + l16) * 72];
      for (int j = 0; j < 8; ++j) {
        vb0[j] = bfbits(vr[kb + j]);      vb0[8 + j] = bfbits(vr[16 + kb + j]);
        vb1[j] = bfbits(vr[32 + kb + j]); vb1[8 + j] = bfbits(vr[48 + kb + j]);
      }
      oacc[nb] = __builtin_amdgcn_wmma_f32_16x16x32_bf16(false, pa[0], false, vb0, (short)0, oacc[nb], false, false);
      oacc[nb] = __builtin_amdgcn_wmma_f32_16x16x32_bf16(false, pa[1], false, vb1, (short)0, oacc[nb], false, false);
    }
    __syncthreads();
  }

  for (int nb = 0; nb < 4; ++nb) {
    const int hd = nb * 16 + l16;
    for (int r = 0; r < 8; ++r) {
      const int g = rowbase + half * 8 + r;
      const float inv = (lrow[r] > 0.f) ? 1.f / lrow[r] : 0.f;
      ao[(long)g * D_DIM + h * HD_N + hd] = oacc[nb][r] * inv;
    }
  }
}

// ---------------------------------------------------------------------------
// out = bf16(silu(a) * b)
// ---------------------------------------------------------------------------
__global__ void silu_mul_kernel(const float* __restrict__ a, const float* __restrict__ b,
                                u16* __restrict__ out, long n) {
  long i = (long)blockIdx.x * blockDim.x + threadIdx.x;
  if (i >= n) return;
  const float x = a[i];
  out[i] = f2bf(x * sigmoidf_(x) * b[i]);
}

// ---------------------------------------------------------------------------
// TTT gradient terms: dgk*l0 and dhk*l2 (bf16) from gk, hk, dhidden, lr
// buffers are [FH, CS, 256]
// ---------------------------------------------------------------------------
__global__ void ttt_grad_kernel(
    const float* __restrict__ gk, const float* __restrict__ hk,
    const float* __restrict__ dh, const float* __restrict__ lr, int chunk,
    u16* __restrict__ dgk_o, u16* __restrict__ dhk_o, long n)
{
  long i = (long)blockIdx.x * blockDim.x + threadIdx.x;
  if (i >= n) return;
  const long te = i >> 8;                 // fh*CS + t
  const int t = (int)(te & (CS_N - 1));
  const int fh = (int)(te >> 10);
  const int s = chunk * CS_N + t;
  const float l0 = lr[(long)s * 12 + 0 * FH_N + fh];
  const float l2 = lr[(long)s * 12 + 2 * FH_N + fh];
  const float g = gk[i], hh = hk[i], d = dh[i];
  const float sg = sigmoidf_(g);
  const float silu = g * sg;
  const float dsilu = sg * (1.f + g * (1.f - sg));
  dhk_o[i] = f2bf(d * silu * l2);
  dgk_o[i] = f2bf(d * hh * dsilu * l0);
}

// ---------------------------------------------------------------------------
// vl1 = bf16(fv * l1) for one chunk; fv [FH,S,FD] -> out [FH,CS,FD]
// ---------------------------------------------------------------------------
__global__ void vl1_kernel(const u16* __restrict__ fv, const float* __restrict__ lr,
                           int chunk, u16* __restrict__ out, long n)
{
  long i = (long)blockIdx.x * blockDim.x + threadIdx.x;
  if (i >= n) return;
  const int d = (int)(i & 255);
  const long tc = i >> 8;
  const int t = (int)(tc & (CS_N - 1));
  const int fh = (int)(tc >> 10);
  const int s = chunk * CS_N + t;
  const float l1 = lr[(long)s * 12 + 1 * FH_N + fh];
  const float v_ = bf2f(fv[((long)fh * S_LEN + s) * FD_N + d]);
  out[i] = f2bf(v_ * l1);
}

// ---------------------------------------------------------------------------
// Final combine: rmsnorm(ttt)*w + ao -> bf16 [S, D]
// ---------------------------------------------------------------------------
__global__ __launch_bounds__(256) void combine_kernel(
    const float* __restrict__ ttt, const float* __restrict__ ao,
    const float* __restrict__ w, u16* __restrict__ out)
{
  __shared__ float red[8];
  const int s = blockIdx.x, fh = blockIdx.y, d = threadIdx.x;
  const float v_ = ttt[((long)fh * S_LEN + s) * FD_N + d];
  float ss = v_ * v_;
  for (int o = 16; o > 0; o >>= 1) ss += __shfl_xor(ss, o, 32);
  if ((d & 31) == 0) red[d >> 5] = ss;
  __syncthreads();
  if (d == 0) { float t = 0.f; for (int i = 0; i < 8; ++i) t += red[i]; red[0] = rsqrtf(t / (float)FD_N + 1e-5f); }
  __syncthreads();
  const float nv = v_ * red[0] * w[d];
  const int col = fh * FD_N + d;
  out[(long)s * D_DIM + col] = f2bf(nv + ao[(long)s * D_DIM + col]);
}

// ---------------------------------------------------------------------------
// Host orchestration
// ---------------------------------------------------------------------------
static void launch_gemm(const u16* A, const u16* B, float* C, int M, int N, int K,
                        int lda, int ldb, int ldc, long sA, long sB, long sC,
                        int tA, int tB, int beta1, int batch, hipStream_t stream) {
  dim3 g(N / 64, M / 64, batch), b(128);
  if (tA == 0 && tB == 0)
    gemm_bf16_wmma<0, 0><<<g, b, 0, stream>>>(A, B, C, M, N, K, lda, ldb, ldc, sA, sB, sC, beta1);
  else if (tA == 0 && tB == 1)
    gemm_bf16_wmma<0, 1><<<g, b, 0, stream>>>(A, B, C, M, N, K, lda, ldb, ldc, sA, sB, sC, beta1);
  else if (tA == 1 && tB == 0)
    gemm_bf16_wmma<1, 0><<<g, b, 0, stream>>>(A, B, C, M, N, K, lda, ldb, ldc, sA, sB, sC, beta1);
  else
    gemm_bf16_wmma<1, 1><<<g, b, 0, stream>>>(A, B, C, M, N, K, lda, ldb, ldc, sA, sB, sC, beta1);
}

extern "C" void kernel_launch(void* const* d_in, const int* in_sizes, int n_in,
                              void* d_out, int out_size, void* d_ws, size_t ws_size,
                              hipStream_t stream) {
  (void)in_sizes; (void)n_in; (void)out_size; (void)ws_size;
  const float* x    = (const float*)d_in[0];
  const float* Wqkv = (const float*)d_in[1];
  const float* qnw  = (const float*)d_in[2];
  const float* knw  = (const float*)d_in[3];
  const float* Wo   = (const float*)d_in[4];
  const float* w0   = (const float*)d_in[5];
  const float* w1   = (const float*)d_in[6];
  const float* w2   = (const float*)d_in[7];
  const float* Wlr  = (const float*)d_in[8];
  const float* blr  = (const float*)d_in[9];
  const float* qksc = (const float*)d_in[10];
  const float* qkof = (const float*)d_in[11];
  const float* tttw = (const float*)d_in[12];
  float* out = (float*)d_out;
  char* ws = (char*)d_ws;

  size_t off = 0;
  auto take = [&](size_t bytes) { size_t r = off; off += (bytes + 255) & ~(size_t)255; return r; };
  const long SD = (long)S_LEN * D_DIM;            // 4M elems
  const long FSD = (long)FH_N * S_LEN * FD_N;     // 4M elems
  const long CHK = (long)FH_N * CS_N * FD_N;      // 1M elems
  const long WSZ = (long)FH_N * 256 * 256;        // 256K elems

  u16*  x_bf     = (u16*)(ws + take(SD * 2));
  u16*  wqkv_bf  = (u16*)(ws + take((long)3 * D_DIM * D_DIM * 2));
  u16*  wo_bf    = (u16*)(ws + take((long)D_DIM * D_DIM * 2));
  float* qkv_f   = (float*)(ws + take((long)S_LEN * 3 * D_DIM * 4));
  float* lr_f    = (float*)(ws + take((long)S_LEN * 12 * 4));
  u16*  fq_bf    = (u16*)(ws + take(FSD * 2));
  u16*  fk_bf    = (u16*)(ws + take(FSD * 2));
  u16*  fv_bf    = (u16*)(ws + take(FSD * 2));
  u16*  Qr_bf    = (u16*)(ws + take(SD * 2));
  u16*  Kr_bf    = (u16*)(ws + take(SD * 2));
  u16*  Vh_bf    = (u16*)(ws + take(SD * 2));
  float* ao_f    = (float*)(ws + take(SD * 4));
  float* W0_f    = (float*)(ws + take(WSZ * 4));
  float* W1_f    = (float*)(ws + take(WSZ * 4));
  float* W2_f    = (float*)(ws + take(WSZ * 4));
  u16*  W0_bf    = (u16*)(ws + take(WSZ * 2));
  u16*  W1_bf    = (u16*)(ws + take(WSZ * 2));
  u16*  W2_bf    = (u16*)(ws + take(WSZ * 2));
  float* gq_f    = (float*)(ws + take(CHK * 4));
  float* hq2_f   = (float*)(ws + take(CHK * 4));
  float* gk_f    = (float*)(ws + take(CHK * 4));
  float* hk_f    = (float*)(ws + take(CHK * 4));
  float* dhid_f  = (float*)(ws + take(CHK * 4));
  u16*  hqprod_bf= (u16*)(ws + take(CHK * 2));
  u16*  hidden_bf= (u16*)(ws + take(CHK * 2));
  u16*  vl1_bf   = (u16*)(ws + take(CHK * 2));
  u16*  dgkl0_bf = (u16*)(ws + take(CHK * 2));
  u16*  dhkl2_bf = (u16*)(ws + take(CHK * 2));
  float* ttt_f   = (float*)(ws + take(FSD * 4));
  u16*  sum_bf   = (u16*)(ws + take(SD * 2));

  auto cvt = [&](const float* in, u16* o_, long n) {
    cvt_f32_bf16_kernel<<<dim3((unsigned)((n + 255) / 256)), dim3(256), 0, stream>>>(in, o_, n);
  };
  auto gemm = [&](const u16* A, const u16* B, float* C, int M, int N, int K,
                  int lda, int ldb, int ldc, long sA, long sB, long sC,
                  int tA, int tB, int beta1, int batch) {
    launch_gemm(A, B, C, M, N, K, lda, ldb, ldc, sA, sB, sC, tA, tB, beta1, batch, stream);
  };

  // ---- projections ----
  cvt(x, x_bf, SD);
  cvt(Wqkv, wqkv_bf, (long)3 * D_DIM * D_DIM);
  cvt(Wo, wo_bf, (long)D_DIM * D_DIM);
  gemm(x_bf, wqkv_bf, qkv_f, S_LEN, 3 * D_DIM, D_DIM,
       D_DIM, D_DIM, 3 * D_DIM, 0, 0, 0, 0, 0, 0, 1);

  rmsnorm_kernel<<<dim3(S_LEN), dim3(256), 0, stream>>>(qkv_f,         qnw, 3 * D_DIM, D_DIM);
  rmsnorm_kernel<<<dim3(S_LEN), dim3(256), 0, stream>>>(qkv_f + D_DIM, knw, 3 * D_DIM, D_DIM);

  lr_kernel<<<dim3(S_LEN), dim3(256), 0, stream>>>(x, Wlr, blr, lr_f);

  // fast-weight q/k/v prep (pre-RoPE q/k)
  fw_kernel<<<dim3(S_LEN, FH_N), dim3(256), 0, stream>>>(qkv_f,             3 * D_DIM, qksc + 0, qkof + 0, 1, fq_bf);
  fw_kernel<<<dim3(S_LEN, FH_N), dim3(256), 0, stream>>>(qkv_f + D_DIM,     3 * D_DIM, qksc + 1, qkof + 1, 1, fk_bf);
  fw_kernel<<<dim3(S_LEN, FH_N), dim3(256), 0, stream>>>(qkv_f + 2 * D_DIM, 3 * D_DIM, (const float*)nullptr, (const float*)nullptr, 0, fv_bf);

  // RoPE + head-major transpose for attention
  {
    const long tot = (long)S_LEN * NH_N * (HD_N / 2);
    const unsigned nb = (unsigned)((tot + 255) / 256);
    rope_transpose_kernel<<<dim3(nb), dim3(256), 0, stream>>>(qkv_f,             Qr_bf, 1, 3 * D_DIM);
    rope_transpose_kernel<<<dim3(nb), dim3(256), 0, stream>>>(qkv_f + D_DIM,     Kr_bf, 1, 3 * D_DIM);
    rope_transpose_kernel<<<dim3(nb), dim3(256), 0, stream>>>(qkv_f + 2 * D_DIM, Vh_bf, 0, 3 * D_DIM);
  }

  // flash sliding-window attention
  attn_kernel<<<dim3(CS_N / 64, NC_N, NH_N), dim3(128), 0, stream>>>(Qr_bf, Kr_bf, Vh_bf, ao_f);

  // ---- TTT scan ----
  (void)hipMemcpyAsync(W0_f, w0, WSZ * 4, hipMemcpyDeviceToDevice, stream);
  (void)hipMemcpyAsync(W1_f, w1, WSZ * 4, hipMemcpyDeviceToDevice, stream);
  (void)hipMemcpyAsync(W2_f, w2, WSZ * 4, hipMemcpyDeviceToDevice, stream);
  cvt(W0_f, W0_bf, WSZ); cvt(W1_f, W1_bf, WSZ); cvt(W2_f, W2_bf, WSZ);

  const long hstride = (long)S_LEN * FD_N;   // per-head stride in fq/fk/fv/ttt
  const long wstride = 256 * 256;            // per-head stride in W*
  const long cstride = (long)CS_N * FD_N;    // per-head stride in chunk temps

  for (int c = 0; c < NC_N; ++c) {
    const u16* qi = fq_bf + (long)c * CS_N * FD_N;
    const u16* ki = fk_bf + (long)c * CS_N * FD_N;
    const u16* vi = fv_bf + (long)c * CS_N * FD_N;

    // apply current fast weights to queries: oi = (silu(q W0^T) * (q W2^T)) W1^T
    gemm(qi, W0_bf, gq_f,  CS_N, 256, 256, FD_N, 256, 256, hstride, wstride, cstride, 0, 0, 0, FH_N);
    gemm(qi, W2_bf, hq2_f, CS_N, 256, 256, FD_N, 256, 256, hstride, wstride, cstride, 0, 0, 0, FH_N);
    silu_mul_kernel<<<dim3((unsigned)((CHK + 255) / 256)), dim3(256), 0, stream>>>(gq_f, hq2_f, hqprod_bf, CHK);
    gemm(hqprod_bf, W1_bf, ttt_f + (long)c * CS_N * FD_N,
         CS_N, 256, 256, 256, 256, 256, cstride, wstride, hstride, 0, 0, 0, FH_N);

    // gradient pieces on keys
    gemm(ki, W0_bf, gk_f, CS_N, 256, 256, FD_N, 256, 256, hstride, wstride, cstride, 0, 0, 0, FH_N);
    gemm(ki, W2_bf, hk_f, CS_N, 256, 256, FD_N, 256, 256, hstride, wstride, cstride, 0, 0, 0, FH_N);
    silu_mul_kernel<<<dim3((unsigned)((CHK + 255) / 256)), dim3(256), 0, stream>>>(gk_f, hk_f, hidden_bf, CHK);
    // dhidden = v @ W1 (W1 transposed at stage)
    gemm(vi, W1_bf, dhid_f, CS_N, 256, 256, FD_N, 256, 256, hstride, wstride, cstride, 0, 1, 0, FH_N);

    ttt_grad_kernel<<<dim3((unsigned)((CHK + 255) / 256)), dim3(256), 0, stream>>>(
        gk_f, hk_f, dhid_f, lr_f, c, dgkl0_bf, dhkl2_bf, CHK);
    vl1_kernel<<<dim3((unsigned)((CHK + 255) / 256)), dim3(256), 0, stream>>>(fv_bf, lr_f, c, vl1_bf, CHK);

    // weight updates (A^T B GEMMs, beta = 1)
    gemm(vl1_bf,   hidden_bf, W1_f, 256, 256, CS_N, 256, 256, 256, cstride, cstride, wstride, 1, 1, 1, FH_N);
    gemm(dgkl0_bf, ki,        W0_f, 256, 256, CS_N, 256, FD_N, 256, cstride, hstride, wstride, 1, 1, 1, FH_N);
    gemm(dhkl2_bf, ki,        W2_f, 256, 256, CS_N, 256, FD_N, 256, cstride, hstride, wstride, 1, 1, 1, FH_N);

    cvt(W0_f, W0_bf, WSZ); cvt(W1_f, W1_bf, WSZ); cvt(W2_f, W2_bf, WSZ);
  }

  // ---- combine + output projection ----
  combine_kernel<<<dim3(S_LEN, FH_N), dim3(256), 0, stream>>>(ttt_f, ao_f, tttw, sum_bf);
  gemm(sum_bf, wo_bf, out, S_LEN, D_DIM, D_DIM, D_DIM, D_DIM, D_DIM, 0, 0, 0, 0, 0, 0, 1);
}
